// CrystalHypergraphConvNet_31387620999325
// MI455X (gfx1250) — matrix operand order repaired and checked by reading, not compiled, not run on hardware
//
#include <hip/hip_runtime.h>
#include <math.h>

// ---------------------------------------------------------------------------
// CrystalHypergraphConvNet for MI455X (gfx1250, wave32, WMMA).
// All dense linear layers run through v_wmma_f32_16x16x32_bf16 (f32 accum).
#define N_ATOMS   30000
#define M_NBR     12
#define ORIG_FEA  92
#define NBR_FEA_C 41
#define TRIP_FEA_C 41
#define MOTIF_FEA_C 59
#define N_TRIP    90000
#define K_TRIP    3
#define N_MOTIF   30000
#define K_MOTIF   8
#define N_CRYS    500
#define ATOM_C    64
#define H_FEA     128
// ---------------------------------------------------------------------------

typedef __attribute__((ext_vector_type(16))) __bf16    bf16x16;
typedef __attribute__((ext_vector_type(8)))  float     f32x8;
typedef __attribute__((ext_vector_type(8)))  unsigned  u32x8;

__device__ __forceinline__ unsigned pack2bf(float lo, float hi) {
  unsigned ul = __builtin_bit_cast(unsigned, lo);
  unsigned uh = __builtin_bit_cast(unsigned, hi);
  ul = (ul + 0x7FFFu + ((ul >> 16) & 1u)) >> 16;          // RNE, low half
  uh = (uh + 0x7FFFu + ((uh >> 16) & 1u)) & 0xFFFF0000u;  // RNE, high half
  return uh | (ul & 0xFFFFu);
}
__device__ __forceinline__ float softplusf(float x) {
  return fmaxf(x, 0.f) + log1pf(expf(-fabsf(x)));
}
__device__ __forceinline__ float sigmoidf(float x) {
  return 1.f / (1.f + expf(-x));
}
// order-preserving float<->uint encoding for atomic max/min
__device__ __forceinline__ unsigned fenc(float f) {
  unsigned u = __builtin_bit_cast(unsigned, f);
  return (u & 0x80000000u) ? ~u : (u | 0x80000000u);
}
__device__ __forceinline__ float fdec(unsigned u) {
  u = (u & 0x80000000u) ? (u & 0x7FFFFFFFu) : ~u;
  return __builtin_bit_cast(float, u);
}

// ---------------------------------------------------------------------------
// GEMM: Y[M,N] = A[M,K] * W[N,K]^T + bias[N]
// Block = 128 threads (4 waves). Block tile 64x64; each wave computes a 16x64
// strip: one A fragment feeds 4 WMMA accumulators per K-step. LDS holds
// pair-packed bf16 tiles (f32->bf16 conversion done at staging), double-
// buffered so global loads of K-tile t+1 overlap WMMA on tile t. Fragment
// fetches are raw ds_load_b128s.
// ---------------------------------------------------------------------------
#define LDSP 20  // padded row stride (dwords): keeps b128 alignment, spreads banks

__global__ __launch_bounds__(128)
void k_gemm_bf16(const float* __restrict__ A, const float* __restrict__ W,
                 const float* __restrict__ bias, float* __restrict__ Y,
                 int M, int K, int N) {
  __shared__ unsigned lA[2][64][LDSP];  // 64 rows x 16 bf16-pairs
  __shared__ unsigned lB[2][64][LDSP];  // 64 out-feature rows x 16 pairs
  const int tid  = threadIdx.x;
  const int lane = tid & 31;
  const int wave = tid >> 5;
  const int m0 = blockIdx.x * 64;
  const int n0 = blockIdx.y * 64;
  const int half = lane >> 4;
  const int lrow = lane & 15;
  const int nk = (K + 31) >> 5;

  f32x8 acc[4];
#pragma unroll
  for (int ns = 0; ns < 4; ++ns) acc[ns] = (f32x8){0.f,0.f,0.f,0.f,0.f,0.f,0.f,0.f};

  // ---- staging (block-uniform fast/slow path selection) ----
  auto stage = [&](int t, int buf) {
    const int k0 = t << 5;
    if (m0 + 64 <= M && k0 + 32 <= K) {        // interior A tile
      for (int i = tid; i < 64 * 16; i += 128) {
        int r = i >> 4, pc = i & 15;
        const float* p = A + (size_t)(m0 + r) * K + k0 + 2 * pc;
        lA[buf][r][pc] = pack2bf(p[0], p[1]);
      }
    } else {
      for (int i = tid; i < 64 * 16; i += 128) {
        int r = i >> 4, pc = i & 15;
        int gm = m0 + r, gk = k0 + 2 * pc;
        float lo = (gm < M && gk < K)     ? A[(size_t)gm * K + gk]     : 0.f;
        float hi = (gm < M && gk + 1 < K) ? A[(size_t)gm * K + gk + 1] : 0.f;
        lA[buf][r][pc] = pack2bf(lo, hi);
      }
    }
    if (n0 + 64 <= N && k0 + 32 <= K) {        // interior W tile
      for (int i = tid; i < 64 * 16; i += 128) {
        int r = i >> 4, pc = i & 15;
        const float* p = W + (size_t)(n0 + r) * K + k0 + 2 * pc;
        lB[buf][r][pc] = pack2bf(p[0], p[1]);
      }
    } else {
      for (int i = tid; i < 64 * 16; i += 128) {
        int r = i >> 4, pc = i & 15;
        int gn = n0 + r, gk = k0 + 2 * pc;
        float lo = (gn < N && gk < K)     ? W[(size_t)gn * K + gk]     : 0.f;
        float hi = (gn < N && gk + 1 < K) ? W[(size_t)gn * K + gk + 1] : 0.f;
        lB[buf][r][pc] = pack2bf(lo, hi);
      }
    }
  };

  stage(0, 0);
  __syncthreads();

  for (int t = 0; t < nk; ++t) {
    const int cur = t & 1;
    if (t + 1 < nk) stage(t + 1, cur ^ 1);     // overlap next tile with WMMA

    // A fragment: two contiguous runs of 4 pairs (b128 each)
    const unsigned* pa = &lA[cur][wave * 16 + lrow][0];
    uint4 a0 = *(const uint4*)(pa + half * 4);
    uint4 a1 = *(const uint4*)(pa + 8 + half * 4);
    u32x8 au; au[0]=a0.x; au[1]=a0.y; au[2]=a0.z; au[3]=a0.w;
              au[4]=a1.x; au[5]=a1.y; au[6]=a1.z; au[7]=a1.w;
    bf16x16 af = __builtin_bit_cast(bf16x16, au);

#pragma unroll
    for (int ns = 0; ns < 4; ++ns) {
      const unsigned* pb = &lB[cur][ns * 16 + lrow][0];
      uint4 b0 = *(const uint4*)(pb + half * 4);
      uint4 b1 = *(const uint4*)(pb + 8 + half * 4);
      u32x8 bu; bu[0]=b0.x; bu[1]=b0.y; bu[2]=b0.z; bu[3]=b0.w;
                bu[4]=b1.x; bu[5]=b1.y; bu[6]=b1.z; bu[7]=b1.w;
      bf16x16 bf_ = __builtin_bit_cast(bf16x16, bu);
      acc[ns] = __builtin_amdgcn_wmma_f32_16x16x32_bf16(false, af, false, bf_,
                                                        (short)0, acc[ns], false, false);
    }
    __syncthreads();
  }

  // D layout: lane -> col (lane&15); vgpr r -> row r + 8*half
#pragma unroll
  for (int ns = 0; ns < 4; ++ns) {
    int n = n0 + ns * 16 + (lane & 15);
    if (n < N) {
      float bv = bias ? bias[n] : 0.f;
#pragma unroll
      for (int r = 0; r < 8; ++r) {
        int m = m0 + wave * 16 + r + 8 * half;
        if (m < M) Y[(size_t)m * N + n] = acc[ns][r] + bv;
      }
    }
  }
}

// ---------------------------------------------------------------------------
// Per-column mean / rsqrt(var+eps) over a row-major [rows, cols] matrix.
// ---------------------------------------------------------------------------
__global__ __launch_bounds__(256)
void k_colstats(const float* __restrict__ Z, int rows, int cols,
                float* __restrict__ mean_o, float* __restrict__ inv_o) {
  __shared__ float s1[256], s2[256];
  const int c = blockIdx.x;
  float s = 0.f, q = 0.f;
  for (int r = threadIdx.x; r < rows; r += blockDim.x) {
    float v = Z[(size_t)r * cols + c];
    s += v; q += v * v;
  }
  s1[threadIdx.x] = s; s2[threadIdx.x] = q;
  __syncthreads();
  for (int st = 128; st > 0; st >>= 1) {
    if (threadIdx.x < st) { s1[threadIdx.x] += s1[threadIdx.x + st];
                            s2[threadIdx.x] += s2[threadIdx.x + st]; }
    __syncthreads();
  }
  if (threadIdx.x == 0) {
    float m = s1[0] / (float)rows;
    float v = s2[0] / (float)rows - m * m;
    mean_o[c] = m;
    inv_o[c]  = rsqrtf(fmaxf(v, 0.f) + 1e-5f);
  }
}

// ---------------------------------------------------------------------------
// conv layer helpers
// ---------------------------------------------------------------------------
__global__ void k_conv_concat(const float* __restrict__ x, const int* __restrict__ nbr,
                              const float* __restrict__ nfea, float* __restrict__ out,
                              int total) {
  int e = blockIdx.x * blockDim.x + threadIdx.x;
  if (e >= total) return;
  int c = e % 169, r = e / 169;
  int i = r / M_NBR;
  float v;
  if (c < 64)       v = x[(size_t)i * 64 + c];
  else if (c < 128) v = x[(size_t)nbr[r] * 64 + (c - 64)];
  else              v = nfea[(size_t)r * NBR_FEA_C + (c - 128)];
  out[e] = v;
}

__global__ void k_conv_gate_sum(const float* __restrict__ Z, const float* __restrict__ mean,
                                const float* __restrict__ inv, const float* __restrict__ g,
                                const float* __restrict__ b, float* __restrict__ out,
                                int n) {
  int idx = blockIdx.x * blockDim.x + threadIdx.x;
  if (idx >= n * 64) return;
  int i = idx >> 6, c = idx & 63;
  float acc = 0.f;
  for (int j = 0; j < M_NBR; ++j) {
    size_t r = (size_t)(i * M_NBR + j) * 128;
    float zf = (Z[r + c]      - mean[c])      * inv[c]      * g[c]      + b[c];
    float zc = (Z[r + 64 + c] - mean[64 + c]) * inv[64 + c] * g[64 + c] + b[64 + c];
    acc += sigmoidf(zf) * softplusf(zc);
  }
  out[idx] = acc;
}

// y = softplus(x + BN(src))
__global__ void k_bn_res_softplus(const float* __restrict__ src, const float* __restrict__ x,
                                  const float* __restrict__ mean, const float* __restrict__ inv,
                                  const float* __restrict__ g, const float* __restrict__ b,
                                  float* __restrict__ y, int total, int C) {
  int idx = blockIdx.x * blockDim.x + threadIdx.x;
  if (idx >= total) return;
  int c = idx % C;
  y[idx] = softplusf(x[idx] + (src[idx] - mean[c]) * inv[c] * g[c] + b[c]);
}

// ---------------------------------------------------------------------------
// chgconv helpers
// ---------------------------------------------------------------------------
__global__ void k_gather(const float* __restrict__ x, const int* __restrict__ hnode,
                         float* __restrict__ xg, int E) {
  int idx = blockIdx.x * blockDim.x + threadIdx.x;
  if (idx >= E * 64) return;
  int e = idx >> 6, c = idx & 63;
  xg[idx] = x[(size_t)hnode[e] * 64 + c];
}

// contiguous fixed-size hyperedge segments -> stacked [H,4,64]
__global__ void k_hedge_stats(const float* __restrict__ xg, float* __restrict__ stk,
                              int H, int kper) {
  int idx = blockIdx.x * blockDim.x + threadIdx.x;
  if (idx >= H * 64) return;
  int h = idx >> 6, c = idx & 63;
  float s = 0.f, q = 0.f, mx = -__builtin_inff(), mn = __builtin_inff();
  for (int m = 0; m < kper; ++m) {
    float v = xg[((size_t)h * kper + m) * 64 + c];
    s += v; q += v * v; mx = fmaxf(mx, v); mn = fminf(mn, v);
  }
  float mean = s / (float)kper;
  float sd = sqrtf(fmaxf(q / (float)kper - mean * mean, 0.f) + 1e-5f);
  size_t base = (size_t)h * 4 * 64 + c;
  stk[base] = mean; stk[base + 64] = sd; stk[base + 128] = mx; stk[base + 192] = mn;
}

// per-(hedge, head) 4x4 attention over stacked tokens; head dim 16
__global__ void k_attn(const float* __restrict__ q, const float* __restrict__ k,
                       const float* __restrict__ v, float* __restrict__ o, int S) {
  int t = blockIdx.x * blockDim.x + threadIdx.x;
  if (t >= S * 4) return;
  int h = t >> 2, head = t & 3;
  size_t base = (size_t)h * 4 * 64 + head * 16;
  float att[4][4];
  for (int s = 0; s < 4; ++s) {
    float mx = -__builtin_inff();
    for (int tt = 0; tt < 4; ++tt) {
      float d = 0.f;
      for (int e = 0; e < 16; ++e)
        d += q[base + s * 64 + e] * k[base + tt * 64 + e];
      att[s][tt] = d * 0.25f;
      mx = fmaxf(mx, att[s][tt]);
    }
    float sum = 0.f;
    for (int tt = 0; tt < 4; ++tt) { att[s][tt] = expf(att[s][tt] - mx); sum += att[s][tt]; }
    float r = 1.f / sum;
    for (int tt = 0; tt < 4; ++tt) att[s][tt] *= r;
  }
  for (int s = 0; s < 4; ++s)
    for (int e = 0; e < 16; ++e) {
      float acc = 0.f;
      for (int tt = 0; tt < 4; ++tt) acc += att[s][tt] * v[base + tt * 64 + e];
      o[base + s * 64 + e] = acc;
    }
}

// y = LN(a + b) (mode 0) or LN(a + relu(b)) (mode 1), rows of 64
__global__ void k_resid_ln(const float* __restrict__ a, const float* __restrict__ b,
                           const float* __restrict__ g, const float* __restrict__ beta,
                           float* __restrict__ y, int rows, int mode) {
  int r = blockIdx.x * blockDim.x + threadIdx.x;
  if (r >= rows) return;
  size_t base = (size_t)r * 64;
  float s = 0.f, q = 0.f;
  for (int c = 0; c < 64; ++c) {
    float bb = b[base + c];
    float v = a[base + c] + (mode ? fmaxf(bb, 0.f) : bb);
    s += v; q += v * v;
  }
  float mean = s * (1.f / 64.f);
  float inv  = rsqrtf(fmaxf(q * (1.f / 64.f) - mean * mean, 0.f) + 1e-5f);
  for (int c = 0; c < 64; ++c) {
    float bb = b[base + c];
    float v = a[base + c] + (mode ? fmaxf(bb, 0.f) : bb);
    y[base + c] = (v - mean) * inv * g[c] + beta[c];
  }
}

__global__ void k_seg_mean4(const float* __restrict__ t, float* __restrict__ out, int S) {
  int idx = blockIdx.x * blockDim.x + threadIdx.x;
  if (idx >= S * 64) return;
  int h = idx >> 6, c = idx & 63;
  size_t b = (size_t)h * 4 * 64 + c;
  out[idx] = 0.25f * (t[b] + t[b + 64] + t[b + 128] + t[b + 192]);
}

__global__ void k_cat2(const float* __restrict__ xg, const float* __restrict__ hm,
                       const float* __restrict__ attr, float* __restrict__ out,
                       int total, int F, int kper) {
  int idx = blockIdx.x * blockDim.x + threadIdx.x;
  if (idx >= total) return;
  int Wc = 128 + F;
  int c = idx % Wc, r = idx / Wc, h = r / kper;
  float v;
  if (c < 64)       v = xg[(size_t)r * 64 + c];
  else if (c < 128) v = hm[(size_t)h * 64 + (c - 64)];
  else              v = attr[(size_t)h * F + (c - 128)];
  out[idx] = v;
}

__global__ void k_gate_chg(const float* __restrict__ z, const float* __restrict__ mean,
                           const float* __restrict__ inv, const float* __restrict__ gf,
                           const float* __restrict__ bf, const float* __restrict__ gc,
                           const float* __restrict__ bc, float* __restrict__ out, int E) {
  int idx = blockIdx.x * blockDim.x + threadIdx.x;
  if (idx >= E * 64) return;
  int e = idx >> 6, c = idx & 63;
  float zf = (z[(size_t)e * 128 + c]      - mean[c])      * inv[c]      * gf[c] + bf[c];
  float zc = (z[(size_t)e * 128 + 64 + c] - mean[64 + c]) * inv[64 + c] * gc[c] + bc[c];
  out[idx] = sigmoidf(zf) * softplusf(zc);
}

// scattered segment stats (node aggregation) via atomics
__global__ void k_seg_init(float* __restrict__ sum, float* __restrict__ sq,
                           unsigned* __restrict__ mx, unsigned* __restrict__ mn,
                           float* __restrict__ cnt, int n) {
  int idx = blockIdx.x * blockDim.x + threadIdx.x;
  if (idx < n * 64) {
    sum[idx] = 0.f; sq[idx] = 0.f;
    mx[idx] = fenc(-__builtin_inff());
    mn[idx] = fenc(__builtin_inff());
  }
  if (idx < n) cnt[idx] = 0.f;
}

__global__ void k_seg_atomic(const float* __restrict__ val, const int* __restrict__ hnode,
                             float* __restrict__ sum, float* __restrict__ sq,
                             unsigned* __restrict__ mx, unsigned* __restrict__ mn,
                             float* __restrict__ cnt, int E) {
  int idx = blockIdx.x * blockDim.x + threadIdx.x;
  if (idx >= E * 64) return;
  int e = idx >> 6, c = idx & 63;
  float v = val[idx];
  size_t o = (size_t)hnode[e] * 64 + c;
  atomicAdd(&sum[o], v);
  atomicAdd(&sq[o], v * v);
  atomicMax(&mx[o], fenc(v));
  atomicMin(&mn[o], fenc(v));
  if (c == 0) atomicAdd(&cnt[hnode[e]], 1.f);
}

__global__ void k_seg_final(const float* __restrict__ sum, const float* __restrict__ sq,
                            const unsigned* __restrict__ mxe, const unsigned* __restrict__ mne,
                            const float* __restrict__ cnt, float* __restrict__ stk, int n) {
  int idx = blockIdx.x * blockDim.x + threadIdx.x;
  if (idx >= n * 64) return;
  int i = idx >> 6, c = idx & 63;
  float ct = fmaxf(cnt[i], 1.f);
  float mean = sum[idx] / ct;
  float sd = sqrtf(fmaxf(sq[idx] / ct - mean * mean, 0.f) + 1e-5f);
  float mx = fdec(mxe[idx]);
  float mn = fdec(mne[idx]);
  if (cnt[i] < 0.5f) { mx = 0.f; mn = 0.f; }
  size_t b = (size_t)i * 4 * 64 + c;
  stk[b] = mean; stk[b + 64] = sd; stk[b + 128] = mx; stk[b + 192] = mn;
}

// ---------------------------------------------------------------------------
// head
// ---------------------------------------------------------------------------
__global__ void k_crys_init(float* __restrict__ sum, float* __restrict__ cnt) {
  int idx = blockIdx.x * blockDim.x + threadIdx.x;
  if (idx < N_CRYS * 64) sum[idx] = 0.f;
  if (idx < N_CRYS) cnt[idx] = 0.f;
}
__global__ void k_crys_atomic(const float* __restrict__ x, const int* __restrict__ cid,
                              float* __restrict__ sum, float* __restrict__ cnt) {
  int idx = blockIdx.x * blockDim.x + threadIdx.x;
  if (idx >= N_ATOMS * 64) return;
  int i = idx >> 6, c = idx & 63;
  atomicAdd(&sum[(size_t)cid[i] * 64 + c], x[idx]);
  if (c == 0) atomicAdd(&cnt[cid[i]], 1.f);
}
__global__ void k_crys_final(const float* __restrict__ sum, const float* __restrict__ cnt,
                             float* __restrict__ out) {
  int idx = blockIdx.x * blockDim.x + threadIdx.x;
  if (idx >= N_CRYS * 64) return;
  out[idx] = softplusf(sum[idx] / fmaxf(cnt[idx >> 6], 1.f));
}
__global__ void k_softplus_inplace(float* __restrict__ y, int total) {
  int idx = blockIdx.x * blockDim.x + threadIdx.x;
  if (idx < total) y[idx] = softplusf(y[idx]);
}

// ---------------------------------------------------------------------------
// host-side orchestration
// ---------------------------------------------------------------------------
static inline void gemm(const float* A, const float* W, const float* b, float* Y,
                        int M, int K, int N, hipStream_t s) {
  dim3 g((M + 63) / 64, (N + 63) / 64);
  k_gemm_bf16<<<g, 128, 0, s>>>(A, W, b, Y, M, K, N);
}
#define EW(n) dim3(((n) + 255) / 256), dim3(256)

// workspace layout (floats). Requires ws_size >= ~1.0 GB.
static constexpr size_t MF   = 1u << 20;
static constexpr size_t XA_O = 0;
static constexpr size_t XB_O = 2 * MF;
static constexpr size_t AP_O = 4 * MF;     // concat GEMM input (<= 360000*169)
static constexpr size_t ZP_O = 66 * MF;    // GEMM z out / gathered xg
static constexpr size_t T0_O = 113 * MF;   // stacked [S,4,64]
static constexpr size_t T1_O = 137 * MF;
static constexpr size_t T2_O = 161 * MF;
static constexpr size_t T3_O = 185 * MF;
static constexpr size_t T4_O = 209 * MF;
static constexpr size_t HM_O = 233 * MF;   // hedge/node message [S,64]
static constexpr size_t SS_O = 239 * MF;
static constexpr size_t SQ_O = 241 * MF;
static constexpr size_t MX_O = 243 * MF;
static constexpr size_t MN_O = 245 * MF;
static constexpr size_t CT_O = 247 * MF;
static constexpr size_t ST_O = 248 * MF;   // bn stats: mean[256], inv[256]
static constexpr size_t CP_O = ST_O + 1024;
static constexpr size_t CC_O = CP_O + 32000;
static constexpr size_t PL_O = CC_O + 512;
static constexpr size_t CH_O = PL_O + 32000;

struct MABP { const float *Wk,*bk,*Wl,*bl,*ln1b,*ln1g,*ln2b,*ln2g,*Wo,*bo,*Wq,*bq,*Wv,*bv; };
struct ChgP { const float *bnc_b,*bnc_g,*bnf_b,*bnf_g,*bno_b,*bno_g,*Wf2,*bf2; MABP hed, nod; };
struct ConvP { const float *bn1b,*bn1g,*bn2b,*bn2g,*W,*b; };

static MABP mab_at(void* const* d, int base) {
  auto L = [&](int i) { return (const float*)d[8 + base + i]; };
  MABP m;
  m.Wk = L(0);  m.bk = L(1);  m.Wl = L(2);  m.bl = L(3);
  m.ln1b = L(4); m.ln1g = L(5); m.ln2b = L(6); m.ln2g = L(7);
  m.Wo = L(8);  m.bo = L(9);  m.Wq = L(10); m.bq = L(11);
  m.Wv = L(12); m.bv = L(13);
  return m;
}
static ChgP chg_at(void* const* d, int base) {
  auto L = [&](int i) { return (const float*)d[8 + base + i]; };
  ChgP p;
  p.bnc_b = L(0); p.bnc_g = L(1); p.bnf_b = L(2); p.bnf_g = L(3);
  p.bno_b = L(4); p.bno_g = L(5);
  p.hed = mab_at(d, base + 6);
  p.Wf2 = L(20); p.bf2 = L(21);
  p.nod = mab_at(d, base + 22);
  return p;
}
static ConvP conv_at(void* const* d, int base) {
  auto L = [&](int i) { return (const float*)d[8 + base + i]; };
  ConvP p; p.bn1b = L(0); p.bn1g = L(1); p.bn2b = L(2); p.bn2g = L(3);
  p.W = L(4); p.b = L(5);
  return p;
}

// MAB on stacked [S,4,64] (t0) -> out [S,64]; t1..t4 scratch
static void run_mab(const MABP& p, float* t0, float* t1, float* t2, float* t3,
                    float* t4, float* out, int S, hipStream_t s) {
  const int R = S * 4;
  gemm(t0, p.Wq, p.bq, t1, R, 64, 64, s);
  gemm(t0, p.Wk, p.bk, t2, R, 64, 64, s);
  gemm(t0, p.Wv, p.bv, t3, R, 64, 64, s);
  k_attn<<<EW(R), 0, s>>>(t1, t2, t3, t4, S);
  gemm(t4, p.Wo, p.bo, t1, R, 64, 64, s);
  k_resid_ln<<<EW(R), 0, s>>>(t1, t0, p.ln1g, p.ln1b, t2, R, 0);
  gemm(t2, p.Wl, p.bl, t3, R, 64, 64, s);
  k_resid_ln<<<EW(R), 0, s>>>(t2, t3, p.ln2g, p.ln2b, t4, R, 1);
  k_seg_mean4<<<EW(S * 64), 0, s>>>(t4, out, S);
}

static void run_chg(const ChgP& p, const int* hnode, const float* attr,
                    int E, int H, int F, int kper,
                    const float* x, float* xnext, float* w, hipStream_t s) {
  float* xg  = w + ZP_O;
  float* t0  = w + T0_O; float* t1 = w + T1_O; float* t2 = w + T2_O;
  float* t3  = w + T3_O; float* t4 = w + T4_O;
  float* hm  = w + HM_O;
  float* mean = w + ST_O; float* inv = w + ST_O + 256;

  k_gather<<<EW(E * 64), 0, s>>>(x, hnode, xg, E);
  k_hedge_stats<<<EW(H * 64), 0, s>>>(xg, t0, H, kper);
  run_mab(p.hed, t0, t1, t2, t3, t4, hm, H, s);

  const int Wc = 128 + F;
  k_cat2<<<EW(E * Wc), 0, s>>>(xg, hm, attr, w + AP_O, E * Wc, F, kper);
  gemm(w + AP_O, p.Wf2, p.bf2, w + ZP_O, E, Wc, 128, s);
  k_colstats<<<128, 256, 0, s>>>(w + ZP_O, E, 128, mean, inv);
  k_gate_chg<<<EW(E * 64), 0, s>>>(w + ZP_O, mean, inv,
                                   p.bnf_g, p.bnf_b, p.bnc_g, p.bnc_b,
                                   w + AP_O, E);

  float* ssum = w + SS_O; float* ssq = w + SQ_O; float* scnt = w + CT_O;
  unsigned* smx = (unsigned*)(w + MX_O); unsigned* smn = (unsigned*)(w + MN_O);
  k_seg_init<<<EW(N_ATOMS * 64), 0, s>>>(ssum, ssq, smx, smn, scnt, N_ATOMS);
  k_seg_atomic<<<EW(E * 64), 0, s>>>(w + AP_O, hnode, ssum, ssq, smx, smn, scnt, E);
  k_seg_final<<<EW(N_ATOMS * 64), 0, s>>>(ssum, ssq, smx, smn, scnt, t0, N_ATOMS);
  run_mab(p.nod, t0, t1, t2, t3, t4, hm, N_ATOMS, s);

  k_colstats<<<64, 256, 0, s>>>(hm, N_ATOMS, 64, mean, inv);
  k_bn_res_softplus<<<EW(N_ATOMS * 64), 0, s>>>(hm, x, mean, inv, p.bno_g, p.bno_b,
                                                xnext, N_ATOMS * 64, 64);
}

static void run_conv(const ConvP& p, const float* nfea, const int* nbr,
                     const float* x, float* xnext, float* w, hipStream_t s) {
  float* mean = w + ST_O; float* inv = w + ST_O + 256;
  const int R = N_ATOMS * M_NBR;
  k_conv_concat<<<EW(R * 169), 0, s>>>(x, nbr, nfea, w + AP_O, R * 169);
  gemm(w + AP_O, p.W, p.b, w + ZP_O, R, 169, 128, s);
  k_colstats<<<128, 256, 0, s>>>(w + ZP_O, R, 128, mean, inv);
  k_conv_gate_sum<<<EW(N_ATOMS * 64), 0, s>>>(w + ZP_O, mean, inv, p.bn1g, p.bn1b,
                                              w + T0_O, N_ATOMS);
  k_colstats<<<64, 256, 0, s>>>(w + T0_O, N_ATOMS, 64, mean, inv);
  k_bn_res_softplus<<<EW(N_ATOMS * 64), 0, s>>>(w + T0_O, x, mean, inv, p.bn2g, p.bn2b,
                                                xnext, N_ATOMS * 64, 64);
}

extern "C" void kernel_launch(void* const* d_in, const int* in_sizes, int n_in,
                              void* d_out, int out_size, void* d_ws, size_t ws_size,
                              hipStream_t stream) {
  (void)in_sizes; (void)n_in; (void)out_size; (void)ws_size;
  const float* atom_fea  = (const float*)d_in[0];
  const float* nbr_fea   = (const float*)d_in[1];
  const int*   nbr_idx   = (const int*)d_in[2];
  const int*   motif_idx = (const int*)d_in[3];   // [2, 240000]
  const float* motif_fea = (const float*)d_in[4];
  const int*   trip_idx  = (const int*)d_in[5];   // [2, 270000]
  const float* trip_fea  = (const float*)d_in[6];
  const int*   crys_idx  = (const int*)d_in[7];
  auto TP = [&](int i) { return (const float*)d_in[8 + i]; };
  const float* cfW = TP(0); const float* cfb = TP(1);   // conv_to_fc
  const float* emW = TP(2); const float* emb = TP(3);   // embedding
  const float* foW = TP(4); const float* fob = TP(5);   // fc_out
  static const int bases[9] = {6, 12, 48, 84, 90, 126, 162, 168, 204};

  float* w = (float*)d_ws;
  float* xa = w + XA_O; float* xb = w + XB_O;

  gemm(atom_fea, emW, emb, xa, N_ATOMS, ORIG_FEA, ATOM_C, stream);
  float* xc = xa; float* xn = xb;

  for (int l = 0; l < 9; ++l) {
    int t = l % 3;
    if (t == 0) {
      ConvP p = conv_at(d_in, bases[l]);
      run_conv(p, nbr_fea, nbr_idx, xc, xn, w, stream);
    } else if (t == 1) {
      ChgP p = chg_at(d_in, bases[l]);
      run_chg(p, trip_idx + N_TRIP * K_TRIP, trip_fea,
              N_TRIP * K_TRIP, N_TRIP, TRIP_FEA_C, K_TRIP, xc, xn, w, stream);
    } else {
      ChgP p = chg_at(d_in, bases[l]);
      run_chg(p, motif_idx + N_MOTIF * K_MOTIF, motif_fea,
              N_MOTIF * K_MOTIF, N_MOTIF, MOTIF_FEA_C, K_MOTIF, xc, xn, w, stream);
    }
    float* tmp = xc; xc = xn; xn = tmp;
  }

  k_crys_init<<<EW(N_CRYS * 64), 0, stream>>>(w + CP_O, w + CC_O);
  k_crys_atomic<<<EW(N_ATOMS * 64), 0, stream>>>(xc, crys_idx, w + CP_O, w + CC_O);
  k_crys_final<<<EW(N_CRYS * 64), 0, stream>>>(w + CP_O, w + CC_O, w + PL_O);
  gemm(w + PL_O, cfW, cfb, w + CH_O, N_CRYS, ATOM_C, H_FEA, stream);
  k_softplus_inplace<<<EW(N_CRYS * H_FEA), 0, stream>>>(w + CH_O, N_CRYS * H_FEA);
  gemm(w + CH_O, foW, fob, (float*)d_out, N_CRYS, H_FEA, 1, stream);
}
